// KernelNNFixed_37297495998594
// MI455X (gfx1250) — matrix operand order repaired and checked by reading, not compiled, not run on hardware
//
#include <hip/hip_runtime.h>
#include <cstdint>
#include <cstddef>

// ---------------- problem constants (from reference) ----------------
#define BN    10000      // batch windows
#define WINT  10         // window steps
#define NNODE 10         // nodes per graph
#define DDIM  3
#define WID   30         // WIDTH = WIN*D
#define KW    64
#define KIN   6
#define NE    150000     // edges
#define NT    100000     // total nodes
#define NPAD  912        // 900 padded to multiple of 16

typedef _Float16 v16h __attribute__((ext_vector_type(16)));
typedef _Float16 v8h  __attribute__((ext_vector_type(8)));
typedef _Float16 h2   __attribute__((ext_vector_type(2)));
typedef float    v8f  __attribute__((ext_vector_type(8)));

__device__ __forceinline__ float sigmoidf_(float x) { return 1.0f / (1.0f + __expf(-x)); }

// ---------------- 0: zero fill ----------------
__global__ void zero_f32(float* __restrict__ p, int n) {
    int i = blockIdx.x * blockDim.x + threadIdx.x;
    if (i < n) p[i] = 0.0f;
}

// ---------------- 1: transpose k2_w into (k, o*30+i) layout, f16, pad to 912 ----------------
__global__ void prep_k2(const float* __restrict__ k2w, const float* __restrict__ k2b,
                        _Float16* __restrict__ k2wT, float* __restrict__ k2bT) {
    int gid = blockIdx.x * blockDim.x + threadIdx.x;
    if (gid >= KW * NPAD) return;
    int k = gid / NPAD, d = gid - k * NPAD;
    _Float16 v = (_Float16)0.0f;
    if (d < 900) {
        int o = d / WID, i = d - o * WID;
        v = (_Float16)k2w[k * 900 + i * WID + o];
    }
    k2wT[gid] = v;
    if (k == 0) {
        float bv = 0.0f;
        if (d < 900) { int o = d / WID, i = d - o * WID; bv = k2b[i * WID + o]; }
        k2bT[d] = bv;
    }
}

// ---------------- 2: fc0 + LSTM, one wave per batch element ----------------
// lane j (<30) owns one channel of h/c.  Output permuted to node-feature layout:
// xfeat[(b*NN + j/3)*30 + t*3 + j%3] = h_t[j]
__global__ void __launch_bounds__(256) lstm_enc(
        const float* __restrict__ xpos, const float* __restrict__ fc0w,
        const float* __restrict__ fc0b, const float* __restrict__ wih,
        const float* __restrict__ whh, const float* __restrict__ bih,
        const float* __restrict__ bhh, float* __restrict__ xfeat) {
    __shared__ float s_f0w[900], s_f0b[WID], s_wih[3600], s_whh[3600], s_bs[120];
    int tid = threadIdx.x;
    for (int i = tid; i < 900;  i += 256) s_f0w[i] = fc0w[i];
    for (int i = tid; i < 3600; i += 256) { s_wih[i] = wih[i]; s_whh[i] = whh[i]; }
    if (tid < WID) s_f0b[tid] = fc0b[tid];
    if (tid < 120) s_bs[tid]  = bih[tid] + bhh[tid];
    __syncthreads();

    int wid = tid >> 5, lane = tid & 31;
    int b = blockIdx.x * 8 + wid;
    if (b >= BN) return;
    int j = (lane < WID) ? lane : (WID - 1);   // clamp so lanes 30/31 stay safe

    float h = 0.0f, c = 0.0f;
    for (int t = 0; t < WINT; ++t) {
        float xr = xpos[(size_t)(b * WINT + t) * WID + j];
        float xt = s_f0b[j];
        #pragma unroll
        for (int i = 0; i < WID; ++i) xt += __shfl(xr, i, 32) * s_f0w[i * WID + j];
        float zi = s_bs[j], zf = s_bs[30 + j], zg = s_bs[60 + j], zo = s_bs[90 + j];
        for (int i = 0; i < WID; ++i) {
            float xb = __shfl(xt, i, 32);
            float hb = __shfl(h,  i, 32);
            const float* wi = s_wih + i * 120;
            const float* wh = s_whh + i * 120;
            zi += xb * wi[j]      + hb * wh[j];
            zf += xb * wi[30 + j] + hb * wh[30 + j];
            zg += xb * wi[60 + j] + hb * wh[60 + j];
            zo += xb * wi[90 + j] + hb * wh[90 + j];
        }
        c = sigmoidf_(zf) * c + sigmoidf_(zi) * tanhf(zg);
        h = sigmoidf_(zo) * tanhf(c);
        if (lane < WID) {
            int n = j / DDIM, d = j - DDIM * n;
            xfeat[(size_t)(b * NNODE + n) * WID + t * DDIM + d] = h;
        }
    }
}

// ---------------- 3: edge kernel-net layers 0,1 (6->64->64, relu), f16 out ----------------
__global__ void __launch_bounds__(128) edge_mlp(
        const float* __restrict__ ea, const float* __restrict__ k0w,
        const float* __restrict__ k0b, const float* __restrict__ k1w,
        const float* __restrict__ k1b, _Float16* __restrict__ hE) {
    __shared__ float s0[KIN * KW], sb0[KW], s1[KW * KW], sb1[KW];
    for (int i = threadIdx.x; i < KIN * KW; i += 128) s0[i] = k0w[i];
    for (int i = threadIdx.x; i < KW;       i += 128) { sb0[i] = k0b[i]; sb1[i] = k1b[i]; }
    for (int i = threadIdx.x; i < KW * KW;  i += 128) s1[i] = k1w[i];
    __syncthreads();
    int e = blockIdx.x * 128 + threadIdx.x;
    if (e >= NE) return;
    float a[KIN];
    #pragma unroll
    for (int i = 0; i < KIN; ++i) a[i] = ea[(size_t)e * KIN + i];
    float h0[KW];
    #pragma unroll
    for (int j = 0; j < KW; ++j) {
        float v = sb0[j];
        #pragma unroll
        for (int i = 0; i < KIN; ++i) v += a[i] * s0[i * KW + j];
        h0[j] = fmaxf(v, 0.0f);
    }
    for (int j = 0; j < KW; ++j) {
        float v = sb1[j];
        #pragma unroll
        for (int i = 0; i < KW; ++i) v += h0[i] * s1[i * KW + j];
        hE[(size_t)e * KW + j] = (_Float16)fmaxf(v, 0.0f);
    }
}

// ---------------- 4: ewT = hE(E x 64, f16) @ k2wT(64 x 912, f16) + bias, f16 out ----------------
// One wave per 16x16 output tile, 4 M-tiles per wave (B operand reused in regs).
// K=64 -> two v_wmma_f32_16x16x32_f16, fp32 accumulate; stores use one base
// address + constant instruction offsets (rows are 1800 B apart).
__global__ void __launch_bounds__(256) ew_gemm(
        const _Float16* __restrict__ hE, const _Float16* __restrict__ k2wT,
        const float* __restrict__ k2bT, _Float16* __restrict__ ewT) {
    const int wid = threadIdx.x >> 5, lane = threadIdx.x & 31;
    const int nt = blockIdx.y;
    const int col = nt * 16 + (lane & 15);        // N index (0..911)
    const int kb2 = (lane < 16) ? 0 : 16;         // B-operand K base per half-wave

    v16h b0, b1;                                  // B for K=0..31 and K=32..63
    #pragma unroll
    for (int j = 0; j < 16; ++j) {
        b0[j] = k2wT[(kb2 + j) * NPAD + col];
        b1[j] = k2wT[(32 + kb2 + j) * NPAD + col];
    }
    const float bias = k2bT[col];

    const int kb = (lane < 16) ? 0 : 16;          // A-operand K base per half-wave
    const int mt0 = (blockIdx.x * 8 + wid) * 4;
    for (int t = 0; t < 4; ++t) {
        int mt = mt0 + t;                         // uniform across the wave
        if (mt >= (NE / 16)) break;
        const _Float16* ar = hE + (size_t)(mt * 16 + (lane & 15)) * KW + kb;
        v8h p0 = *(const v8h*)(ar);               // K kb..kb+7
        v8h p1 = *(const v8h*)(ar + 8);           // K kb+8..kb+15
        v8h p2 = *(const v8h*)(ar + 32);          // K 32+kb..
        v8h p3 = *(const v8h*)(ar + 40);          // K 40+kb..
        v16h a0, a1;
        #pragma unroll
        for (int j = 0; j < 8; ++j) { a0[j] = p0[j]; a0[8 + j] = p1[j];
                                      a1[j] = p2[j]; a1[8 + j] = p3[j]; }
        v8f c = {};
        c = __builtin_amdgcn_wmma_f32_16x16x32_f16(false, a0, false, b0, (short)0, c, false, false);
        c = __builtin_amdgcn_wmma_f32_16x16x32_f16(false, a1, false, b1, (short)0, c, false, false);
        if (col < 900) {
            int rbase = mt * 16 + ((lane < 16) ? 0 : 8);
            _Float16* outp = ewT + (size_t)rbase * 900 + col;   // single address calc
            #pragma unroll
            for (int r = 0; r < 8; ++r)
                outp[r * 900] = (_Float16)(c[r] + bias);        // folds into imm offsets
        }
    }
}

// ---------------- 5: in-degree counts ----------------
__global__ void count_deg(const int* __restrict__ ei, float* __restrict__ cnt) {
    int e = blockIdx.x * blockDim.x + threadIdx.x;
    if (e < NE) atomicAdd(&cnt[ei[NE + e]], 1.0f);
}

// ---------------- 6: per-edge message + scatter-add ----------------
// One wave per edge; lane o computes msg[o] = sum_i x[src,i]*ewT[e, o*30+i]
__global__ void __launch_bounds__(256) conv_msg(
        const float* __restrict__ x, const _Float16* __restrict__ ewT,
        const int* __restrict__ ei, float* __restrict__ agg) {
    int wid = threadIdx.x >> 5, lane = threadIdx.x & 31;
    int e = blockIdx.x * 8 + wid;
    if (e >= NE) return;
    int src = ei[e], dst = ei[NE + e];
    int o = (lane < WID) ? lane : (WID - 1);
    float xs = x[(size_t)src * WID + o];
    const _Float16* w = ewT + (size_t)e * 900 + o * WID;  // 4-byte aligned (60*o + 1800*e)
    float msg = 0.0f;
    #pragma unroll
    for (int j = 0; j < 15; ++j) {
        h2 p = *(const h2*)(w + 2 * j);
        msg += __shfl(xs, 2 * j, 32) * (float)p[0] + __shfl(xs, 2 * j + 1, 32) * (float)p[1];
    }
    if (lane < WID) atomicAdd(&agg[(size_t)dst * WID + o], msg);
}

// ---------------- 7: node update  y = relu(agg/cnt + x@root + bias) ----------------
__global__ void __launch_bounds__(256) node_update(
        const float* __restrict__ x, const float* __restrict__ agg,
        const float* __restrict__ cnt, const float* __restrict__ root,
        const float* __restrict__ bias, float* __restrict__ y) {
    __shared__ float s_r[900], s_b[WID];
    for (int i = threadIdx.x; i < 900; i += 256) s_r[i] = root[i];
    if (threadIdx.x < WID) s_b[threadIdx.x] = bias[threadIdx.x];
    __syncthreads();
    int gid = blockIdx.x * 256 + threadIdx.x;
    if (gid >= NT * WID) return;
    int n = gid / WID, o = gid - n * WID;
    const float* xr = x + (size_t)n * WID;
    float acc = 0.0f;
    #pragma unroll
    for (int i = 0; i < WID; ++i) acc += xr[i] * s_r[i * WID + o];
    float cc = fmaxf(cnt[n], 1.0f);
    y[gid] = fmaxf(agg[gid] / cc + acc + s_b[o], 0.0f);
}

// ---------------- 8: final fc3 (no relu) ----------------
__global__ void __launch_bounds__(256) fc3_k(
        const float* __restrict__ x, const float* __restrict__ w,
        const float* __restrict__ b, float* __restrict__ y) {
    __shared__ float s_w[900], s_b[WID];
    for (int i = threadIdx.x; i < 900; i += 256) s_w[i] = w[i];
    if (threadIdx.x < WID) s_b[threadIdx.x] = b[threadIdx.x];
    __syncthreads();
    int gid = blockIdx.x * 256 + threadIdx.x;
    if (gid >= NT * WID) return;
    int n = gid / WID, o = gid - n * WID;
    const float* xr = x + (size_t)n * WID;
    float acc = 0.0f;
    #pragma unroll
    for (int i = 0; i < WID; ++i) acc += xr[i] * s_w[i * WID + o];
    y[gid] = acc + s_b[o];
}

// =====================================================================
extern "C" void kernel_launch(void* const* d_in, const int* in_sizes, int n_in,
                              void* d_out, int out_size, void* d_ws, size_t ws_size,
                              hipStream_t stream) {
    const float* xpos  = (const float*)d_in[0];
    const int*   ei    = (const int*)  d_in[1];
    const float* eattr = (const float*)d_in[2];
    const float* fc0w  = (const float*)d_in[3];
    const float* fc0b  = (const float*)d_in[4];
    const float* wih   = (const float*)d_in[5];
    const float* whh   = (const float*)d_in[6];
    const float* bih   = (const float*)d_in[7];
    const float* bhh   = (const float*)d_in[8];
    const float* k0w   = (const float*)d_in[9];
    const float* k0b   = (const float*)d_in[10];
    const float* k1w   = (const float*)d_in[11];
    const float* k1b   = (const float*)d_in[12];
    const float* k2w   = (const float*)d_in[13];
    const float* k2b   = (const float*)d_in[14];
    const float* root1 = (const float*)d_in[15];
    const float* bias1 = (const float*)d_in[16];
    const float* root2 = (const float*)d_in[17];
    const float* bias2 = (const float*)d_in[18];
    const float* fc3w  = (const float*)d_in[19];
    const float* fc3b  = (const float*)d_in[20];
    float* out = (float*)d_out;

    // workspace carve-up (256B aligned)
    char* ws = (char*)d_ws;
    size_t off = 0;
    auto carve = [&](size_t bytes) { char* p = ws + off; off = (off + bytes + 255) & ~(size_t)255; return p; };
    float*    xA   = (float*)   carve((size_t)NT * WID * 4);   // 12 MB
    float*    xB   = (float*)   carve((size_t)NT * WID * 4);   // 12 MB
    float*    agg  = (float*)   carve((size_t)NT * WID * 4);   // 12 MB
    float*    cnt  = (float*)   carve((size_t)NT * 4);         // 0.4 MB
    _Float16* hE   = (_Float16*)carve((size_t)NE * KW * 2);    // 19.2 MB
    _Float16* k2wT = (_Float16*)carve((size_t)KW * NPAD * 2);  // 117 KB
    float*    k2bT = (float*)   carve((size_t)NPAD * 4);       // 3.6 KB
    _Float16* ewT  = (_Float16*)carve((size_t)NE * 900 * 2);   // 270 MB

    // 1) weight prep
    prep_k2<<<(KW * NPAD + 255) / 256, 256, 0, stream>>>(k2w, k2b, k2wT, k2bT);
    // 2) temporal encoder -> node features
    lstm_enc<<<(BN + 7) / 8, 256, 0, stream>>>(xpos, fc0w, fc0b, wih, whh, bih, bhh, xA);
    // 3) edge MLP -> h (f16)
    edge_mlp<<<(NE + 127) / 128, 128, 0, stream>>>(eattr, k0w, k0b, k1w, k1b, hE);
    // 4) edge-conditioned weights (WMMA), stored transposed + f16
    ew_gemm<<<dim3((NE / 16 + 31) / 32, NPAD / 16), 256, 0, stream>>>(hE, k2wT, k2bT, ewT);
    // 5) degree counts
    zero_f32<<<(NT + 255) / 256, 256, 0, stream>>>(cnt, NT);
    count_deg<<<(NE + 255) / 256, 256, 0, stream>>>(ei, cnt);

    // 6) 4 x NNConv
    float* xcur = xA; float* xnxt = xB;
    for (int conv = 0; conv < 4; ++conv) {
        const float* root = (conv < 2) ? root1 : root2;
        const float* bias = (conv < 2) ? bias1 : bias2;
        zero_f32<<<(NT * WID + 255) / 256, 256, 0, stream>>>(agg, NT * WID);
        conv_msg<<<(NE + 7) / 8, 256, 0, stream>>>(xcur, ewT, ei, agg);
        node_update<<<(NT * WID + 255) / 256, 256, 0, stream>>>(xcur, agg, cnt, root, bias, xnxt);
        float* t = xcur; xcur = xnxt; xnxt = t;
    }
    // 7) fc3 -> output
    fc3_k<<<(NT * WID + 255) / 256, 256, 0, stream>>>(xcur, fc3w, fc3b, out);
}
